// HybridBlock_64682207478028
// MI455X (gfx1250) — matrix operand order repaired and checked by reading, not compiled, not run on hardware
//
#include <hip/hip_runtime.h>
#include <hip/hip_bf16.h>

// ---------------------------------------------------------------------------
// Problem constants (match reference):  N=20000, E=320000, HID=256, H=4, C=64
// ---------------------------------------------------------------------------
#define HID   256
#define NHEAD 4
#define HDIM  64

typedef __attribute__((ext_vector_type(16))) __bf16 v16bf;
typedef __attribute__((ext_vector_type(8)))  __bf16 v8bf;
typedef __attribute__((ext_vector_type(8)))  float  v8f;

// ---- ordered-uint encoding for float atomic max --------------------------
__device__ __forceinline__ unsigned int enc_ord(float f) {
    unsigned int u = __float_as_uint(f);
    return (u & 0x80000000u) ? ~u : (u | 0x80000000u);
}
__device__ __forceinline__ float dec_ord(unsigned int e) {
    return (e & 0x80000000u) ? __uint_as_float(e ^ 0x80000000u)
                             : __uint_as_float(~e);
}
#define ENC_NEG_INF 0x007FFFFFu   // enc_ord(-inf)

// ---------------------------------------------------------------------------
// Weight convert + transpose: dst[o*in + i] = bf16(src[i*out + o])
// ---------------------------------------------------------------------------
__global__ __launch_bounds__(256)
void transpose_bf16_kernel(__bf16* __restrict__ dst, const float* __restrict__ src,
                           int in_dim, int out_dim) {
    int idx = blockIdx.x * 256 + threadIdx.x;
    if (idx >= in_dim * out_dim) return;
    int o = idx / in_dim;
    int i = idx - o * in_dim;
    dst[idx] = (__bf16)src[(size_t)i * out_dim + o];
}

__global__ __launch_bounds__(256)
void pack_bias_kernel(float* __restrict__ b_all, const float* __restrict__ bq,
                      const float* __restrict__ bk, const float* __restrict__ bv,
                      const float* __restrict__ bs) {
    int i = blockIdx.x * 256 + threadIdx.x;
    if (i >= 4 * HID) return;
    int m = i >> 8, c = i & 255;
    const float* s = (m == 0) ? bq : (m == 1) ? bk : (m == 2) ? bv : bs;
    b_all[i] = s[c];
}

// ---------------------------------------------------------------------------
// Init scratch: agg=0, mmax=enc(-inf), den=0   (re-run every call: replay-safe)
// ---------------------------------------------------------------------------
__global__ __launch_bounds__(256)
void init_kernel(float* __restrict__ agg, unsigned int* __restrict__ mmax,
                 float* __restrict__ den, int n_nodes) {
    int idx = blockIdx.x * 256 + threadIdx.x;
    if (idx < n_nodes * HID) agg[idx] = 0.0f;
    if (idx < n_nodes * NHEAD) { mmax[idx] = ENC_NEG_INF; den[idx] = 0.0f; }
}

// ---------------------------------------------------------------------------
// LayerNorm over HID=256, one node per block (256 thr = 8 waves), bf16 out
// ---------------------------------------------------------------------------
__global__ __launch_bounds__(256)
void ln_kernel(const float* __restrict__ x, const float* __restrict__ w,
               const float* __restrict__ b, __bf16* __restrict__ out) {
    __shared__ float red[8];
    const int node = blockIdx.x;
    const int tid  = threadIdx.x;
    float v = x[(size_t)node * HID + tid];

    float s = v;
    for (int m = 16; m >= 1; m >>= 1) s += __shfl_xor(s, m, 32);
    if ((tid & 31) == 0) red[tid >> 5] = s;
    __syncthreads();
    float tot = 0.f;
#pragma unroll
    for (int i = 0; i < 8; ++i) tot += red[i];
    float mu = tot * (1.0f / HID);
    __syncthreads();

    float d  = v - mu;
    float s2 = d * d;
    for (int m = 16; m >= 1; m >>= 1) s2 += __shfl_xor(s2, m, 32);
    if ((tid & 31) == 0) red[tid >> 5] = s2;
    __syncthreads();
    float var = 0.f;
#pragma unroll
    for (int i = 0; i < 8; ++i) var += red[i];
    var *= (1.0f / HID);

    float y = d * rsqrtf(var + 1e-5f) * w[tid] + b[tid];
    out[(size_t)node * HID + tid] = (__bf16)y;
}

// ---------------------------------------------------------------------------
// WMMA bf16 GEMM with explicit double-buffered software pipeline.
//  out[M,Ncols] = A[M,K](bf16,row-major) * BT[Ncols,K](bf16)^T + bias
//  Each wave accumulates a 16 x (16*NT) strip; fragments for k-step i+1 are
//  loaded while the WMMAs for k-step i execute, so the s_wait before each
//  WMMA group only covers loads issued one iteration earlier (partial waits,
//  loads stay in flight).  Requires K % 64 == 0 (true here: 256/512).
//  EPI 0: f32 out   EPI 1: exact GELU -> bf16 out   EPI 2: +residual -> f32
// Fragment layouts used here (wave32; contiguous k order, not the ISA table order):
//  A: lane(0-15)=row M, khalf=lane>>4; elems j -> K=k0+16*khalf+j (A contiguous)
//  B: lane col = lane&15, elems j -> K = k0 + 16*(lane>>4) + j  (BT contiguous)
//  C: elem r -> row = m0 + r + 8*(lane>>4), col = n0 + (lane&15)
// ---------------------------------------------------------------------------
template <int NT>
struct FragSet {
    v8bf a_lo, a_hi;
    v8bf b_lo[NT], b_hi[NT];
};

template <int NT>
__device__ __forceinline__ void load_frags(FragSet<NT>& f, const __bf16* aptr,
                                           const __bf16* bptr, int K, int k0) {
    f.a_lo = *(const v8bf*)(aptr + k0);
    f.a_hi = *(const v8bf*)(aptr + k0 + 8);
#pragma unroll
    for (int t = 0; t < NT; ++t) {
        const __bf16* bp = bptr + (size_t)16 * t * K + k0;
        f.b_lo[t] = *(const v8bf*)(bp);
        f.b_hi[t] = *(const v8bf*)(bp + 8);
    }
}

template <int NT>
__device__ __forceinline__ void mma_frags(const FragSet<NT>& f, v8f acc[NT]) {
    v16bf a;
#pragma unroll
    for (int j = 0; j < 8; ++j) { a[j] = f.a_lo[j]; a[8 + j] = f.a_hi[j]; }
#pragma unroll
    for (int t = 0; t < NT; ++t) {
        v16bf b;
#pragma unroll
        for (int j = 0; j < 8; ++j) { b[j] = f.b_lo[t][j]; b[8 + j] = f.b_hi[t][j]; }
        acc[t] = __builtin_amdgcn_wmma_f32_16x16x32_bf16(
            false, a, false, b, (short)0, acc[t], false, false);
    }
}

template <int EPI, int NT>
__global__ __launch_bounds__(256)
void wmma_gemm_kernel(const __bf16* __restrict__ A, const __bf16* __restrict__ BT,
                      const float* __restrict__ bias, const float* __restrict__ residual,
                      void* __restrict__ outv, int M, int K, int Ncols) {
    const int lane   = threadIdx.x & 31;
    const int wave   = threadIdx.x >> 5;
    const int strips = Ncols / (16 * NT);
    const int strip  = blockIdx.y * 8 + wave;
    if (strip >= strips) return;
    const int m0    = blockIdx.x * 16;
    const int khalf = lane >> 4;
    const int rlo   = lane & 15;
    const int arow  = m0 + rlo;
    const int bcol0 = strip * 16 * NT + rlo;   // column of this lane in tile 0

    const __bf16* aptr = A + (size_t)arow * K + khalf * 16;          // + k0
    const __bf16* bptr = BT + (size_t)bcol0 * K + khalf * 16;        // + 16*t*K + k0

    v8f acc[NT];
#pragma unroll
    for (int t = 0; t < NT; ++t) acc[t] = (v8f){};

    // -------- software pipeline: ping-pong fragment sets ------------------
    FragSet<NT> f0, f1;
    load_frags<NT>(f0, aptr, bptr, K, 0);
    int k0 = 0;
    for (; k0 + 128 <= K; k0 += 64) {
        load_frags<NT>(f1, aptr, bptr, K, k0 + 32);   // in flight during...
        mma_frags<NT>(f0, acc);                       // ...these WMMAs
        load_frags<NT>(f0, aptr, bptr, K, k0 + 64);
        mma_frags<NT>(f1, acc);
    }
    // tail: 64 K-values remain, f0 holds k0
    load_frags<NT>(f1, aptr, bptr, K, k0 + 32);
    mma_frags<NT>(f0, acc);
    mma_frags<NT>(f1, acc);

    // -------- epilogue ----------------------------------------------------
#pragma unroll
    for (int t = 0; t < NT; ++t) {
        const int bcol = bcol0 + 16 * t;
        const float bb = bias[bcol];
#pragma unroll
        for (int r = 0; r < 8; ++r) {
            int row = m0 + r + 8 * khalf;
            if (row >= M) continue;
            size_t oidx = (size_t)row * Ncols + bcol;
            float v = acc[t][r] + bb;
            if (EPI == 0) {
                ((float*)outv)[oidx] = v;
            } else if (EPI == 1) {
                float g = 0.5f * v * (1.0f + erff(v * 0.70710678118654752f));
                ((__bf16*)outv)[oidx] = (__bf16)g;
            } else {
                ((float*)outv)[oidx] = v + residual[oidx];
            }
        }
    }
}

// ---------------------------------------------------------------------------
// Edge kernels. qkvs row layout: [0,256)=q  [256,512)=k  [512,768)=v  [768,1024)=skip
// One wave per edge: lane covers 8 contiguous floats; lanes [8h,8h+8) = head h.
// ---------------------------------------------------------------------------
__global__ __launch_bounds__(256)
void edge_scores_kernel(const int* __restrict__ ei, const float* __restrict__ qkvs,
                        float* __restrict__ scores, unsigned int* __restrict__ mmax,
                        int E) {
    const int wave = threadIdx.x >> 5;
    const int lane = threadIdx.x & 31;
    const int e    = blockIdx.x * 8 + wave;
    if (e >= E) return;
    const int src = ei[e];
    const int dst = ei[E + e];
    const float* qr = qkvs + (size_t)dst * (4 * HID) + lane * 8;          // q
    const float* kr = qkvs + (size_t)src * (4 * HID) + HID + lane * 8;    // k
    float4 q0 = *(const float4*)qr, q1 = *(const float4*)(qr + 4);
    float4 k0 = *(const float4*)kr, k1 = *(const float4*)(kr + 4);
    float p = q0.x*k0.x + q0.y*k0.y + q0.z*k0.z + q0.w*k0.w
            + q1.x*k1.x + q1.y*k1.y + q1.z*k1.z + q1.w*k1.w;
    p += __shfl_xor(p, 1, 32);
    p += __shfl_xor(p, 2, 32);
    p += __shfl_xor(p, 4, 32);
    const int head = lane >> 3;
    if ((lane & 7) == 0) {
        float sc = p * 0.125f;                         // 1/sqrt(64)
        scores[(size_t)e * NHEAD + head] = sc;
        atomicMax(&mmax[dst * NHEAD + head], enc_ord(sc));
    }
}

__global__ __launch_bounds__(256)
void edge_exp_kernel(const int* __restrict__ ei, float* __restrict__ scores,
                     const unsigned int* __restrict__ mmax, float* __restrict__ den,
                     int E) {
    int idx = blockIdx.x * 256 + threadIdx.x;
    if (idx >= E * NHEAD) return;
    int e = idx >> 2, h = idx & 3;
    int dst = ei[E + e];
    float m = dec_ord(mmax[dst * NHEAD + h]);
    float p = expf(scores[idx] - m);
    scores[idx] = p;                                   // in-place: scores -> exp
    atomicAdd(&den[dst * NHEAD + h], p);
}

__global__ __launch_bounds__(256)
void edge_agg_kernel(const int* __restrict__ ei, const float* __restrict__ qkvs,
                     const float* __restrict__ scores, const float* __restrict__ den,
                     float* __restrict__ agg, int E) {
    const int wave = threadIdx.x >> 5;
    const int lane = threadIdx.x & 31;
    const int e    = blockIdx.x * 8 + wave;
    if (e >= E) return;
    const int src  = ei[e];
    const int dst  = ei[E + e];
    const int head = lane >> 3;
    float attn = scores[(size_t)e * NHEAD + head] / den[dst * NHEAD + head];
    const float* vr = qkvs + (size_t)src * (4 * HID) + 2 * HID + lane * 8;  // v
    float4 v0 = *(const float4*)vr, v1 = *(const float4*)(vr + 4);
    float* op = agg + (size_t)dst * HID + lane * 8;
    atomicAdd(op + 0, attn * v0.x); atomicAdd(op + 1, attn * v0.y);
    atomicAdd(op + 2, attn * v0.z); atomicAdd(op + 3, attn * v0.w);
    atomicAdd(op + 4, attn * v1.x); atomicAdd(op + 5, attn * v1.y);
    atomicAdd(op + 6, attn * v1.z); atomicAdd(op + 7, attn * v1.w);
}

// x1 = x + agg + skip
__global__ __launch_bounds__(256)
void combine_kernel(const float* __restrict__ x, const float* __restrict__ agg,
                    const float* __restrict__ qkvs, float* __restrict__ x1, int n_nodes) {
    int idx = blockIdx.x * 256 + threadIdx.x;
    if (idx >= n_nodes * HID) return;
    int row = idx >> 8, col = idx & 255;
    x1[idx] = x[idx] + agg[idx] + qkvs[(size_t)row * (4 * HID) + 3 * HID + col];
}

// ---------------------------------------------------------------------------
extern "C" void kernel_launch(void* const* d_in, const int* in_sizes, int n_in,
                              void* d_out, int out_size, void* d_ws, size_t ws_size,
                              hipStream_t stream) {
    const float* x     = (const float*)d_in[0];
    const int*   ei    = (const int*)d_in[1];
    const float* ln1_w = (const float*)d_in[2];
    const float* ln1_b = (const float*)d_in[3];
    const float* ln2_w = (const float*)d_in[4];
    const float* ln2_b = (const float*)d_in[5];
    const float* wq    = (const float*)d_in[6];
    const float* bq    = (const float*)d_in[7];
    const float* wk    = (const float*)d_in[8];
    const float* bk    = (const float*)d_in[9];
    const float* wv    = (const float*)d_in[10];
    const float* bv    = (const float*)d_in[11];
    const float* wsk   = (const float*)d_in[12];
    const float* bsk   = (const float*)d_in[13];
    const float* w1    = (const float*)d_in[14];
    const float* b1    = (const float*)d_in[15];
    const float* w2    = (const float*)d_in[16];
    const float* b2    = (const float*)d_in[17];
    float* out = (float*)d_out;

    const int N = in_sizes[0] / HID;        // 20000
    const int E = in_sizes[1] / 2;          // 320000

    // ---- workspace layout (256B aligned regions) -------------------------
    char* ws = (char*)d_ws;
    size_t off = 0;
    auto take = [&](size_t bytes) { char* p = ws + off; off = (off + bytes + 255) & ~(size_t)255; return p; };
    __bf16* wt_all = (__bf16*)take((size_t)4 * HID * HID * 2);     // [1024][256] bf16
    __bf16* w1t    = (__bf16*)take((size_t)2 * HID * HID * 2);     // [512][256]  bf16
    __bf16* w2t    = (__bf16*)take((size_t)2 * HID * HID * 2);     // [256][512]  bf16
    float*  b_all  = (float*)take((size_t)4 * HID * 4);            // [1024] f32
    __bf16* hbuf   = (__bf16*)take((size_t)N * HID * 2);           // LN out (reused LN1/LN2)
    float*  qkvs   = (float*)take((size_t)N * 4 * HID * 4);        // [N][1024] f32
    float*  scores = (float*)take((size_t)E * NHEAD * 4);          // [E][4] f32 (scores->exp)
    unsigned int* mmax = (unsigned int*)take((size_t)N * NHEAD * 4);
    float*  den    = (float*)take((size_t)N * NHEAD * 4);
    float*  agg    = (float*)take((size_t)N * HID * 4);            // reused as ff1 bf16 [N][512]
    float*  x1     = (float*)take((size_t)N * HID * 4);
    __bf16* ff1    = (__bf16*)agg;                                 // same bytes, free after combine
    (void)ws_size;

    // ---- 1) weights -> bf16 transposed ----------------------------------
    transpose_bf16_kernel<<<256, 256, 0, stream>>>(wt_all + 0 * HID * HID, wq,  HID, HID);
    transpose_bf16_kernel<<<256, 256, 0, stream>>>(wt_all + 1 * HID * HID, wk,  HID, HID);
    transpose_bf16_kernel<<<256, 256, 0, stream>>>(wt_all + 2 * HID * HID, wv,  HID, HID);
    transpose_bf16_kernel<<<256, 256, 0, stream>>>(wt_all + 3 * HID * HID, wsk, HID, HID);
    transpose_bf16_kernel<<<512, 256, 0, stream>>>(w1t, w1, HID, 2 * HID);      // [512][256]
    transpose_bf16_kernel<<<512, 256, 0, stream>>>(w2t, w2, 2 * HID, HID);      // [256][512]
    pack_bias_kernel<<<4, 256, 0, stream>>>(b_all, bq, bk, bv, bsk);

    // ---- 2) init scratch -------------------------------------------------
    init_kernel<<<(N * HID + 255) / 256, 256, 0, stream>>>(agg, mmax, den, N);

    // ---- 3) LN1 -> bf16 --------------------------------------------------
    ln_kernel<<<N, 256, 0, stream>>>(x, ln1_w, ln1_b, hbuf);

    // ---- 4) fused q|k|v|skip GEMM  [N,256]x[256,1024] --------------------
    //      16 strips of 16x64 -> grid.y = 2
    wmma_gemm_kernel<0, 4><<<dim3(N / 16, 2), 256, 0, stream>>>(
        hbuf, wt_all, b_all, nullptr, qkvs, N, HID, 4 * HID);

    // ---- 5) edge attention ----------------------------------------------
    edge_scores_kernel<<<(E + 7) / 8, 256, 0, stream>>>(ei, qkvs, scores, mmax, E);
    edge_exp_kernel<<<(E * NHEAD + 255) / 256, 256, 0, stream>>>(ei, scores, mmax, den, E);
    edge_agg_kernel<<<(E + 7) / 8, 256, 0, stream>>>(ei, qkvs, scores, den, agg, E);

    // ---- 6) residual + skip ---------------------------------------------
    combine_kernel<<<(N * HID + 255) / 256, 256, 0, stream>>>(x, agg, qkvs, x1, N);

    // ---- 7) LN2 -> bf16 (reuse hbuf) ------------------------------------
    ln_kernel<<<N, 256, 0, stream>>>(x1, ln2_w, ln2_b, hbuf);

    // ---- 8) FFN1 + exact GELU -> bf16 (reuse agg bytes) ------------------
    //      8 strips of 16x64 -> grid.y = 1
    wmma_gemm_kernel<1, 4><<<dim3(N / 16, 1), 256, 0, stream>>>(
        hbuf, w1t, b1, nullptr, ff1, N, HID, 2 * HID);

    // ---- 9) FFN2 + residual -> d_out  (NT=2 keeps all 8 waves active) ----
    wmma_gemm_kernel<2, 2><<<dim3(N / 16, 1), 256, 0, stream>>>(
        ff1, w2t, b2, x1, out, N, 2 * HID, HID);
}
